// MoeBERTScorer_78675210928168
// MI455X (gfx1250) — matrix operand order repaired and checked by reading, not compiled, not run on hardware
//
#include <hip/hip_runtime.h>
#include <math.h>

// ---------------------------------------------------------------------------
// MoE-BERT forward on gfx1250 (MI455X). Compute-bound on WMMA: all GEMMs go
// through V_WMMA_F32_16X16X32_BF16 (bf16 A/B, f32 accum). fp32 operands are
// converted to bf16 (RNE) once while staging tiles into LDS as packed bf16;
// fragments are then direct dword LDS loads (no VALU packing before WMMA).
// ---------------------------------------------------------------------------

typedef __bf16          v16bf __attribute__((ext_vector_type(16)));
typedef unsigned        v8u   __attribute__((ext_vector_type(8)));
typedef float           v8f   __attribute__((ext_vector_type(8)));

__device__ __forceinline__ unsigned short f2bf(float f) {
    unsigned u = __builtin_bit_cast(unsigned, f);
    u += 0x7FFFu + ((u >> 16) & 1u);          // round-to-nearest-even
    return (unsigned short)(u >> 16);
}

// Generic batched GEMM: C[z] = alpha * A[z] (MxK, row major, lda) *
//                               B[z]  (KxN, element (k,n) at k*brs + n*bcs)
//                               + bias[n]
// batch: z1 = z/zdiv, z2 = z%zdiv; operand offset = z1*S1 + z2*S2.
// Block = 128 threads (4 waves) computes a 32(M) x 64(N) tile of C.
// Each wave owns one 16-wide N slice and TWO 16x16 accumulators stacked in M
// (B-fragment reuse x2, A-tile reuse x4). A[32x32] and B[32x64] tiles are
// staged in LDS as packed bf16, row stride 17 dwords (odd -> conflict-free,
// K-pairs dword-aligned). All M used here are multiples of 32.
__global__ void __launch_bounds__(128)
gemm_bf16_wmma(const float* __restrict__ A, int lda, long long aS1, long long aS2,
               const float* __restrict__ Bm, int brs, int bcs,
               long long bS1, long long bS2,
               const float* __restrict__ bias,
               float* __restrict__ C, int ldc, long long cS1, long long cS2,
               int M, int N, int K, float alpha, int zdiv)
{
    __shared__ unsigned Alds[32 * 17];    // [m][k/2] packed bf16 pairs
    __shared__ unsigned Blds[64 * 17];    // [n][k/2] packed bf16 pairs

    unsigned short* A16 = (unsigned short*)Alds;
    unsigned short* B16 = (unsigned short*)Blds;

    const int tid  = threadIdx.x;
    const int wave = tid >> 5;
    const int lane = tid & 31;
    const int tileM  = blockIdx.y * 32;
    const int tileN0 = blockIdx.x * 64;

    const int z  = blockIdx.z;
    const int z1 = z / zdiv, z2 = z % zdiv;
    A  += (long long)z1 * aS1 + (long long)z2 * aS2;
    Bm += (long long)z1 * bS1 + (long long)z2 * bS2;
    C  += (long long)z1 * cS1 + (long long)z2 * cS2;

    const int half = lane >> 4;
    const int l16  = lane & 15;
    const int nloc = wave * 16 + l16;        // 0..63 within block tile
    const int ncol = tileN0 + nloc;
    const int Nc   = N - 1;

    const bool kfast = (brs == 1);           // B contiguous along k?

    v8f acc0 = {0.f, 0.f, 0.f, 0.f, 0.f, 0.f, 0.f, 0.f};   // rows tileM..+15
    v8f acc1 = {0.f, 0.f, 0.f, 0.f, 0.f, 0.f, 0.f, 0.f};   // rows tileM+16..+31

    for (int k0 = 0; k0 < K; k0 += 32) {
        // ---- stage A tile [32 m x 32 k] -> bf16: 1024 elems, 8/thread -----
#pragma unroll
        for (int i = 0; i < 8; ++i) {
            const int e  = tid + i * 128;
            const int mr = e >> 5, kc = e & 31;
            A16[mr * 34 + kc] = f2bf(A[(tileM + mr) * lda + (k0 + kc)]);
        }
        // ---- stage B tile [32 k x 64 n] -> B16[n][k]: 2048 elems ----------
#pragma unroll
        for (int i = 0; i < 16; ++i) {
            const int e = tid + i * 128;
            int kr, nc;
            if (kfast) { kr = e & 31;  nc = e >> 5; }   // contiguous along k
            else       { nc = e & 63;  kr = e >> 6; }   // contiguous along n
            int gcol = tileN0 + nc;
            if (gcol > Nc) gcol = Nc;                   // clamp (N=7 case)
            B16[nc * 34 + kr] = f2bf(Bm[(k0 + kr) * brs + gcol * bcs]);
        }
        __syncthreads();

        // ---- fragments (ISA 7.12.2): packed K-pairs, direct dword loads ---
        v8u au0, au1, bu;
#pragma unroll
        for (int v = 0; v < 8; ++v) {
            const int kap = ((v & 4) ? 8 : 0) + half * 4 + (v & 3);  // ka/2
            au0[v] = Alds[l16 * 17 + kap];
            au1[v] = Alds[(16 + l16) * 17 + kap];
            const int kbp = ((v & 4) ? 8 : 0) + half * 4 + (v & 3);  // kb/2
            bu[v]  = Blds[nloc * 17 + kbp];
        }
        const v16bf bb16 = __builtin_bit_cast(v16bf, bu);
        acc0 = __builtin_amdgcn_wmma_f32_16x16x32_bf16(
                   false, __builtin_bit_cast(v16bf, au0),
                   false, bb16, (short)0, acc0, false, false);
        acc1 = __builtin_amdgcn_wmma_f32_16x16x32_bf16(
                   false, __builtin_bit_cast(v16bf, au1),
                   false, bb16, (short)0, acc1, false, false);
        __syncthreads();
    }

    if (ncol < N) {
        const float bb = bias ? bias[ncol] : 0.f;
#pragma unroll
        for (int r = 0; r < 8; ++r) {
            const int mm = tileM + r + half * 8;       // C layout: m = r + 8*half
            C[(size_t)mm * ldc + ncol]        = acc0[r] * alpha + bb;
            C[(size_t)(mm + 16) * ldc + ncol] = acc1[r] * alpha + bb;
        }
    }
}

// --- embedding gather: emb = word[id] + pos[s] + type[tt] -------------------
__global__ void embed_kernel(const int* __restrict__ ids, const int* __restrict__ tts,
                             const float* __restrict__ we, const float* __restrict__ pe,
                             const float* __restrict__ te, float* __restrict__ out)
{
    const int i = blockIdx.x * blockDim.x + threadIdx.x;   // [0, 2048*768)
    const int h = i % 768;
    const int t = i / 768;
    const int s = t % 512;
    const int id = ids[t];
    const int ty = tts[t];
    out[i] = we[(size_t)id * 768 + h] + pe[(size_t)s * 768 + h] + te[(size_t)ty * 768 + h];
}

// --- LayerNorm over H=768 with optional residual, one block per token -------
__global__ void __launch_bounds__(256)
ln_kernel(const float* __restrict__ x, const float* __restrict__ res,
          const float* __restrict__ g, const float* __restrict__ b,
          float* __restrict__ out)
{
    __shared__ float red[256];
    const int t = blockIdx.x, tid = threadIdx.x;
    const float* xp = x + (size_t)t * 768;
    const float* rp = res ? res + (size_t)t * 768 : nullptr;
    float v0 = xp[tid]       + (rp ? rp[tid]       : 0.f);
    float v1 = xp[tid + 256] + (rp ? rp[tid + 256] : 0.f);
    float v2 = xp[tid + 512] + (rp ? rp[tid + 512] : 0.f);

    red[tid] = v0 + v1 + v2; __syncthreads();
    for (int o = 128; o > 0; o >>= 1) { if (tid < o) red[tid] += red[tid + o]; __syncthreads(); }
    const float mean = red[0] / 768.f; __syncthreads();

    const float d0 = v0 - mean, d1 = v1 - mean, d2 = v2 - mean;
    red[tid] = d0 * d0 + d1 * d1 + d2 * d2; __syncthreads();
    for (int o = 128; o > 0; o >>= 1) { if (tid < o) red[tid] += red[tid + o]; __syncthreads(); }
    const float inv = rsqrtf(red[0] / 768.f + 1e-12f);

    float* op = out + (size_t)t * 768;
    op[tid]       = d0 * inv * g[tid]       + b[tid];
    op[tid + 256] = d1 * inv * g[tid + 256] + b[tid + 256];
    op[tid + 512] = d2 * inv * g[tid + 512] + b[tid + 512];
}

// --- masked row softmax over key dim (512), one block per (b,h,q) row -------
__global__ void __launch_bounds__(256)
softmax_kernel(float* __restrict__ sc, const int* __restrict__ mask)
{
    __shared__ float red[256];
    const int row = blockIdx.x;                 // B*NH*S rows
    const int b   = row / (12 * 512);
    const int tid = threadIdx.x;
    float* p = sc + (size_t)row * 512;
    const int* mk = mask + b * 512;

    float x0 = p[tid]       + (1.f - (float)mk[tid])       * -10000.f;
    float x1 = p[tid + 256] + (1.f - (float)mk[tid + 256]) * -10000.f;

    red[tid] = fmaxf(x0, x1); __syncthreads();
    for (int o = 128; o > 0; o >>= 1) { if (tid < o) red[tid] = fmaxf(red[tid], red[tid + o]); __syncthreads(); }
    const float mx = red[0]; __syncthreads();

    x0 = __expf(x0 - mx); x1 = __expf(x1 - mx);
    red[tid] = x0 + x1; __syncthreads();
    for (int o = 128; o > 0; o >>= 1) { if (tid < o) red[tid] += red[tid + o]; __syncthreads(); }
    const float inv = 1.f / red[0];

    p[tid] = x0 * inv;
    p[tid + 256] = x1 * inv;
}

// --- exact GELU -------------------------------------------------------------
__global__ void gelu_kernel(float* __restrict__ x, int n)
{
    const int i = blockIdx.x * blockDim.x + threadIdx.x;
    if (i < n) {
        const float v = x[i];
        x[i] = 0.5f * v * (1.f + erff(v * 0.70710678118654752f));
    }
}

// --- top-2 of 7 gate softmax, one thread per token --------------------------
__global__ void gate_kernel(const float* __restrict__ gl, float* __restrict__ gates, int T)
{
    const int t = blockIdx.x * blockDim.x + threadIdx.x;
    if (t >= T) return;
    float g[7];
#pragma unroll
    for (int e = 0; e < 7; ++e) g[e] = gl[t * 7 + e];
    int i1 = 0;
    for (int e = 1; e < 7; ++e) if (g[e] > g[i1]) i1 = e;
    int i2 = (i1 == 0) ? 1 : 0;
    for (int e = 0; e < 7; ++e) if (e != i1 && g[e] > g[i2]) i2 = e;
    const float m  = fmaxf(g[i1], g[i2]);
    const float e1 = __expf(g[i1] - m), e2 = __expf(g[i2] - m);
    const float inv = 1.f / (e1 + e2);
#pragma unroll
    for (int e = 0; e < 7; ++e) gates[t * 7 + e] = 0.f;
    gates[t * 7 + i1] = e1 * inv;
    gates[t * 7 + i2] = e2 * inv;
}

// --- moe accumulate: moe (+)= gates[:,e] * eo -------------------------------
__global__ void moe_acc_kernel(const float* __restrict__ eo, const float* __restrict__ gates,
                               float* __restrict__ moe, int e, int n)
{
    const int i = blockIdx.x * blockDim.x + threadIdx.x;
    if (i >= n) return;
    const int t = i / 768;
    const float v = gates[t * 7 + e] * eo[i];
    moe[i] = (e == 0) ? v : moe[i] + v;
}

// --- masked mean pool + regression head, one block per batch ----------------
__global__ void __launch_bounds__(256)
pool_score_kernel(const float* __restrict__ hid, const int* __restrict__ mask,
                  const float* __restrict__ rw, const float* __restrict__ rb,
                  float* __restrict__ out)
{
    __shared__ float red[256];
    const int b = blockIdx.x, tid = threadIdx.x;
    float msum = 0.f;
    for (int s = 0; s < 512; ++s) msum += (float)mask[b * 512 + s];
    msum = fmaxf(msum, 1e-6f);
    float part = 0.f;
    for (int j = 0; j < 3; ++j) {
        const int h = tid + j * 256;
        float acc = 0.f;
        for (int s = 0; s < 512; ++s)
            acc += hid[((size_t)b * 512 + s) * 768 + h] * (float)mask[b * 512 + s];
        part += (acc / msum) * rw[h];
    }
    red[tid] = part; __syncthreads();
    for (int o = 128; o > 0; o >>= 1) { if (tid < o) red[tid] += red[tid + o]; __syncthreads(); }
    if (tid == 0) out[b] = red[0] + rb[0];
}

// ---------------------------------------------------------------------------
static void gemm(hipStream_t s,
                 const float* A, int lda, long long aS1, long long aS2,
                 const float* B, int brs, int bcs, long long bS1, long long bS2,
                 const float* bias,
                 float* C, int ldc, long long cS1, long long cS2,
                 int M, int N, int K, float alpha, int zdiv, int nz)
{
    dim3 g((N + 63) / 64, M / 32, nz);    // all M here are multiples of 32
    gemm_bf16_wmma<<<g, 128, 0, s>>>(A, lda, aS1, aS2, B, brs, bcs, bS1, bS2,
                                     bias, C, ldc, cS1, cS2, M, N, K, alpha, zdiv);
}

extern "C" void kernel_launch(void* const* d_in, const int* in_sizes, int n_in,
                              void* d_out, int out_size, void* d_ws, size_t ws_size,
                              hipStream_t stream)
{
    (void)in_sizes; (void)n_in; (void)out_size; (void)ws_size;

    const int* ids   = (const int*)d_in[0];
    const int* amask = (const int*)d_in[1];
    const int* tti   = (const int*)d_in[2];
    const float* we   = (const float*)d_in[3];
    const float* pe   = (const float*)d_in[4];
    const float* te   = (const float*)d_in[5];
    const float* elnw = (const float*)d_in[6];
    const float* elnb = (const float*)d_in[7];
    const float* qw   = (const float*)d_in[8];
    const float* qb   = (const float*)d_in[9];
    const float* kw   = (const float*)d_in[10];
    const float* kb   = (const float*)d_in[11];
    const float* vw   = (const float*)d_in[12];
    const float* vb   = (const float*)d_in[13];
    const float* ow   = (const float*)d_in[14];
    const float* ob   = (const float*)d_in[15];
    const float* alnw = (const float*)d_in[16];
    const float* alnb = (const float*)d_in[17];
    const float* gw   = (const float*)d_in[18];
    const float* gb   = (const float*)d_in[19];
    const float* w1   = (const float*)d_in[20];
    const float* b1   = (const float*)d_in[21];
    const float* w2   = (const float*)d_in[22];
    const float* b2   = (const float*)d_in[23];
    const float* olnw = (const float*)d_in[24];
    const float* olnb = (const float*)d_in[25];
    const float* rw   = (const float*)d_in[26];
    const float* rbia = (const float*)d_in[27];

    float* out = (float*)d_out;
    float* ws  = (float*)d_ws;

    const int T = 2048, HH = 768, II = 3072;
    const size_t TH = (size_t)T * HH;

    size_t off = 0;
    float* hid   = ws + off; off += TH;
    float* qbuf  = ws + off; off += TH;
    float* kbuf  = ws + off; off += TH;
    float* vbuf  = ws + off; off += TH;
    float* ctx   = ws + off; off += TH;
    float* tmp   = ws + off; off += TH;
    float* attn  = ws + off; off += TH;
    float* moe   = ws + off; off += TH;
    float* eo    = ws + off; off += TH;
    float* sc    = ws + off; off += (size_t)4 * 12 * 512 * 512;
    float* mid   = ws + off; off += (size_t)T * II;
    float* gl    = ws + off; off += (size_t)T * 7;
    float* gates = ws + off; off += (size_t)T * 7;

    // embedding + LN
    embed_kernel<<<(T * HH) / 256, 256, 0, stream>>>(ids, tti, we, pe, te, tmp);
    ln_kernel<<<T, 256, 0, stream>>>(tmp, nullptr, elnw, elnb, hid);

    const long long bSH = (long long)512 * HH;     // per-batch stride in [B,S,H]
    const long long sBH = (long long)12 * 512 * 512;
    const long long sH  = (long long)512 * 512;

    for (int l = 0; l < 4; ++l) {
        // Q, K, V projections: [2048,768] @ [768,768] + bias
        gemm(stream, hid, HH, 0, 0, qw + (size_t)l * HH * HH, HH, 1, 0, 0,
             qb + l * HH, qbuf, HH, 0, 0, T, HH, HH, 1.f, 1, 1);
        gemm(stream, hid, HH, 0, 0, kw + (size_t)l * HH * HH, HH, 1, 0, 0,
             kb + l * HH, kbuf, HH, 0, 0, T, HH, HH, 1.f, 1, 1);
        gemm(stream, hid, HH, 0, 0, vw + (size_t)l * HH * HH, HH, 1, 0, 0,
             vb + l * HH, vbuf, HH, 0, 0, T, HH, HH, 1.f, 1, 1);

        // scores[b,h] = Q[b,:,h*64:(h+1)*64] @ K^T / 8  -> 48 batches (z/12=b, z%12=h)
        gemm(stream, qbuf, HH, bSH, 64,
             kbuf, 1, HH, bSH, 64,
             nullptr, sc, 512, sBH, sH,
             512, 512, 64, 0.125f, 12, 48);
        softmax_kernel<<<4 * 12 * 512, 256, 0, stream>>>(sc, amask);

        // ctx[b,h] = P @ V  -> 48 batches
        gemm(stream, sc, 512, sBH, sH,
             vbuf, HH, 1, bSH, 64,
             nullptr, ctx, HH, bSH, 64,
             512, 64, 512, 1.f, 12, 48);

        // output projection + residual LN
        gemm(stream, ctx, HH, 0, 0, ow + (size_t)l * HH * HH, HH, 1, 0, 0,
             ob + l * HH, tmp, HH, 0, 0, T, HH, HH, 1.f, 1, 1);
        ln_kernel<<<T, 256, 0, stream>>>(tmp, hid, alnw + l * HH, alnb + l * HH, attn);

        // gating logits [2048,7] + top-2 softmax
        gemm(stream, attn, HH, 0, 0, gw + (size_t)l * HH * 7, 7, 1, 0, 0,
             gb + l * 7, gl, 7, 0, 0, T, 7, HH, 1.f, 1, 1);
        gate_kernel<<<(T + 255) / 256, 256, 0, stream>>>(gl, gates, T);

        // dense all-expert FFN, expert loop (reuses mid/eo buffers)
        for (int e = 0; e < 7; ++e) {
            const float* W1 = w1 + ((size_t)l * 7 + e) * HH * II;
            const float* B1 = b1 + ((size_t)l * 7 + e) * II;
            const float* W2 = w2 + ((size_t)l * 7 + e) * II * HH;
            const float* B2 = b2 + ((size_t)l * 7 + e) * HH;
            gemm(stream, attn, HH, 0, 0, W1, II, 1, 0, 0, B1,
                 mid, II, 0, 0, T, II, HH, 1.f, 1, 1);
            gelu_kernel<<<(T * II) / 256, 256, 0, stream>>>(mid, T * II);
            gemm(stream, mid, II, 0, 0, W2, HH, 1, 0, 0, B2,
                 eo, HH, 0, 0, T, HH, II, 1.f, 1, 1);
            moe_acc_kernel<<<(T * HH) / 256, 256, 0, stream>>>(eo, gates, moe, e, T * HH);
        }

        // output LN (residual = attn_out), write next hidden in place
        ln_kernel<<<T, 256, 0, stream>>>(moe, attn, olnw + l * HH, olnb + l * HH, hid);
    }

    // outputs: score[4], hidden[2048*768], last-layer gates[2048*7]
    pool_score_kernel<<<4, 256, 0, stream>>>(hid, amask, rw, rbia, out);
    hipMemcpyAsync(out + 4, hid, TH * sizeof(float), hipMemcpyDeviceToDevice, stream);
    hipMemcpyAsync(out + 4 + TH, gates, (size_t)T * 7 * sizeof(float),
                   hipMemcpyDeviceToDevice, stream);
}
